// FlowNetC_29454885716571
// MI455X (gfx1250) — matrix operand order repaired and checked by reading, not compiled, not run on hardware
//
#include <hip/hip_runtime.h>
#include <math.h>

// ---------------------------------------------------------------------------
// CDNA5 WMMA (wave32): bf16 A/B fragments (16 elems/lane, 8 VGPRs),
// f32 C/D accumulator (8 elems/lane).
// ---------------------------------------------------------------------------
typedef __bf16 bf16_t;
typedef bf16_t v16bf __attribute__((ext_vector_type(16)));
typedef float  v8f   __attribute__((ext_vector_type(8)));

__device__ __forceinline__ unsigned short f2bf(float f) {
  unsigned int u = __float_as_uint(f);
  unsigned int r = u + 0x7FFFu + ((u >> 16) & 1u);
  return (unsigned short)(r >> 16);
}

__device__ __forceinline__ void wait_async0() {
#if defined(__has_builtin)
#if __has_builtin(__builtin_amdgcn_s_wait_asynccnt)
  __builtin_amdgcn_s_wait_asynccnt(0);
#else
  asm volatile("s_wait_asynccnt 0" ::: "memory");
#endif
#else
  asm volatile("s_wait_asynccnt 0" ::: "memory");
#endif
}

// ---------------------------------------------------------------------------
// Generic batched WMMA GEMM.
//  A: fp32, gathered (dense / conv-im2col / deconv-im2col), converted to bf16
//     into LDS.
//  B: bf16 packed n-major [N][ldk], ldk % 32 == 0, zero-padded beyond K.
//     Staged into LDS with global_load_async_to_lds_b128 (ASYNCcnt).
//  Epilogue: alpha, bias, BN scale/shift x2, lrelu/gelu, residual,
//  column-offset store (in-place channel concat). Activations NHWC fp32.
// ---------------------------------------------------------------------------
struct GemmP {
  const float* A;
  const unsigned short* Bm;   // bf16 bits, n-major [N][ldk]
  float* C;
  const float* bias;
  const float* scale1; const float* shift1;
  const float* scale2; const float* shift2;
  const float* resid;
  long long bAstride, bBstride, bCstride;
  int M, N, K;
  int lda, ldb, ldc;          // ldb = ldk (padded K pitch)
  int colOff;
  float alpha;
  int act;    // 0 none, 1 lrelu(0.1), 2 exact gelu
  int amode;  // 0 dense, 1 conv im2col, 2 deconv (lhs_dilation=2) im2col
  int Cin, H, W, KH, KW, stride, pad, Ho, Wo;
};

union FragBF { v16bf v; unsigned int u[8]; };

__global__ __launch_bounds__(256)
void wmma_gemm_k(GemmP p) {
  __shared__ unsigned short As[128][34];   // +2 pad to stagger banks
  __shared__ unsigned short Bsm[64][32];   // n-major tile; rows 64B, 16B-aligned

  const int tid  = threadIdx.x;
  const int lane = tid & 31;
  const int wave = tid >> 5;
  const int wm = wave >> 1;       // 0..3 (M)
  const int wn = wave & 1;        // 0..1 (N)
  const int m0 = blockIdx.y * 128;
  const int n0 = blockIdx.x * 64;
  const long long bz = blockIdx.z;

  // ---- A loader: thread stages row ar, 16 cols from acb ----
  const int ar  = tid >> 1;
  const int acb = (tid & 1) * 16;
  const int am  = m0 + ar;
  const bool arow_ok = (am < p.M);
  int gb = 0, goy = 0, gox = 0;
  if (p.amode != 0) {
    int hw = p.Ho * p.Wo;
    int mm = arow_ok ? am : 0;
    gb = mm / hw; int rem = mm - gb * hw;
    goy = rem / p.Wo; gox = rem - goy * p.Wo;
  }
  // ---- B loader: thread stages n-row brow, 8 contiguous k from bko ----
  const int brow = tid >> 2;      // 0..63
  const int bko  = (tid & 3) * 8;
  const bool brow_ok = (n0 + brow < p.N);
  const unsigned short* bbase =
      p.Bm + bz * p.bBstride + (long long)(n0 + brow) * p.ldb + bko;
  const unsigned ldsB = (unsigned)(size_t)&Bsm[brow][bko];

  v8f acc[2][2];
#pragma unroll
  for (int i = 0; i < 2; i++)
#pragma unroll
    for (int j = 0; j < 2; j++)
#pragma unroll
      for (int e = 0; e < 8; e++) acc[i][j][e] = 0.0f;

  const int khalf = lane >> 4;
  const int l15   = lane & 15;

  for (int k0 = 0; k0 < p.K; k0 += 32) {
    // ------- stage A tile (gather + fp32->bf16 via VALU) -------
    {
      int kbase = k0 + acb;
      if (!arow_ok) {
#pragma unroll
        for (int i = 0; i < 16; i++) As[ar][acb + i] = 0;
      } else if (p.amode == 0) {
        const float* ap = p.A + bz * p.bAstride + (long long)am * p.lda + kbase;
#pragma unroll 4
        for (int i = 0; i < 16; i++) {
          float f = (kbase + i < p.K) ? ap[i] : 0.0f;
          As[ar][acb + i] = f2bf(f);
        }
      } else {
        // incremental (ci,kx,ky) odometer: one div/mod chain per tile
        int ci = kbase % p.Cin;
        int rr = kbase / p.Cin;
        int kx = rr % p.KW;
        int ky = rr / p.KW;
        const float* abase = p.A + (long long)gb * p.H * p.W * p.Cin;
#pragma unroll 4
        for (int i = 0; i < 16; i++) {
          float f = 0.0f;
          if (kbase + i < p.K) {
            if (p.amode == 1) {
              int iy = goy * p.stride - p.pad + ky;
              int ix = gox * p.stride - p.pad + kx;
              if (iy >= 0 && iy < p.H && ix >= 0 && ix < p.W)
                f = abase[((long long)iy * p.W + ix) * p.Cin + ci];
            } else {
              int iyd = goy - p.pad + ky;
              int ixd = gox - p.pad + kx;
              if (iyd >= 0 && ixd >= 0 && !(iyd & 1) && !(ixd & 1)) {
                int iy = iyd >> 1, ix = ixd >> 1;
                if (iy < p.H && ix < p.W)
                  f = abase[((long long)iy * p.W + ix) * p.Cin + ci];
              }
            }
          }
          As[ar][acb + i] = f2bf(f);
          if (++ci == p.Cin) { ci = 0; if (++kx == p.KW) { kx = 0; ++ky; } }
        }
      }
    }
    // ------- stage B tile: async DMA global -> LDS (16B per lane) -------
    // ldk % 32 == 0 and zero-padded, so k-range is always in bounds;
    // rows with n >= N are skipped (their columns are discarded later).
    if (brow_ok) {
      unsigned long long ga = (unsigned long long)(size_t)(bbase + k0);
      asm volatile("global_load_async_to_lds_b128 %0, %1, off"
                   :: "v"(ldsB), "v"(ga)
                   : "memory");
    }
    wait_async0();
    __syncthreads();

    // ------- build fragments per ISA 7.12.2 layouts -------
    FragBF afr[2], bfr[2];
#pragma unroll
    for (int tm = 0; tm < 2; tm++) {
      int arow = wm * 32 + tm * 16 + l15;
#pragma unroll
      for (int v4 = 0; v4 < 4; v4++) {
        int ka = khalf * 8 + 2 * v4;
        afr[tm].u[v4]     = *(const unsigned int*)&As[arow][ka];
        afr[tm].u[4 + v4] = *(const unsigned int*)&As[arow][16 + ka];
      }
    }
#pragma unroll
    for (int tn = 0; tn < 2; tn++) {
      int bcol = wn * 32 + tn * 16 + l15;
#pragma unroll
      for (int v4 = 0; v4 < 8; v4++) {
        int kb = (v4 >> 2) * 16 + khalf * 8 + 2 * (v4 & 3);
        bfr[tn].u[v4] = *(const unsigned int*)&Bsm[bcol][kb];
      }
    }
    // ------- 4x v_wmma_f32_16x16x32_bf16 -------
#pragma unroll
    for (int tm = 0; tm < 2; tm++)
#pragma unroll
      for (int tn = 0; tn < 2; tn++)
        acc[tm][tn] = __builtin_amdgcn_wmma_f32_16x16x32_bf16(
            false, afr[tm].v, false, bfr[tn].v, (short)0, acc[tm][tn],
            false, false);
    __syncthreads();
  }

  // ------- epilogue -------
#pragma unroll
  for (int tm = 0; tm < 2; tm++)
#pragma unroll
    for (int tn = 0; tn < 2; tn++)
#pragma unroll
      for (int v = 0; v < 8; v++) {
        int row = m0 + wm * 32 + tm * 16 + khalf * 8 + v;
        int col = n0 + wn * 32 + tn * 16 + l15;
        if (row < p.M && col < p.N) {
          float val = acc[tm][tn][v] * p.alpha;
          if (p.bias)   val += p.bias[col];
          if (p.scale1) val = val * p.scale1[col] + p.shift1[col];
          if (p.act == 1)      val = (val >= 0.0f) ? val : 0.1f * val;
          else if (p.act == 2) val = 0.5f * val * (1.0f + erff(val * 0.70710678118654752f));
          if (p.scale2) val = val * p.scale2[col] + p.shift2[col];
          long long idx = bz * p.bCstride + (long long)row * p.ldc + p.colOff + col;
          if (p.resid) val += p.resid[idx];
          p.C[idx] = val;
        }
      }
}

// ---------------------------------------------------------------------------
// Support kernels
// ---------------------------------------------------------------------------
__global__ void nchw_to_nhwc_k(const float* src, float* dst, int B, int Ctot,
                               int H, int W, int c0, int C) {
  long long i = (long long)blockIdx.x * 256 + threadIdx.x;
  long long n = (long long)B * H * W * C;
  if (i >= n) return;
  int c = (int)(i % C); long long r = i / C;
  int x = (int)(r % W); r /= W;
  int y = (int)(r % H); int b = (int)(r / H);
  dst[i] = src[(((long long)b * Ctot + c0 + c) * H + y) * W + x];
}

__global__ void nhwc_to_nchw_k(const float* src, float* dst, int B, int C,
                               int H, int W) {
  long long i = (long long)blockIdx.x * 256 + threadIdx.x;
  long long n = (long long)B * C * H * W;
  if (i >= n) return;
  int x = (int)(i % W); long long r = i / W;
  int y = (int)(r % H); r /= H;
  int c = (int)(r % C); int b = (int)(r / C);
  dst[i] = src[(((long long)b * H + y) * W + x) * C + c];
}

__global__ void bnfold_k(const float* g, const float* b, const float* m,
                         const float* v, float* scale, float* shift, int C) {
  int c = blockIdx.x * 256 + threadIdx.x;
  if (c >= C) return;
  float s = g[c] * rsqrtf(v[c] + 1e-5f);
  scale[c] = s;
  shift[c] = b[c] - m[c] * s;
}

// W [K,N] row-major (or batched [Bn,K,N]) -> n-major bf16 [Bn][N][ldk], 0-pad.
__global__ void pack_w_tr_k(const float* src, unsigned short* dst, int K,
                            int N, int ldk, long long sstride, int Bn) {
  long long i = (long long)blockIdx.x * 256 + threadIdx.x;
  long long tot = (long long)Bn * N * ldk;
  if (i >= tot) return;
  int k = (int)(i % ldk); long long r = i / ldk;
  int n = (int)(r % N); int b = (int)(r / N);
  float f = (k < K) ? src[b * sstride + (long long)k * N + n] : 0.0f;
  dst[i] = f2bf(f);
}

// src row-major [Bn][N,K] -> n-major bf16 [Bn][N][ldk], 0-pad. (B = src^T)
__global__ void pack_w_direct_k(const float* src, unsigned short* dst, int K,
                                int N, int ldk, long long sstride, int Bn) {
  long long i = (long long)blockIdx.x * 256 + threadIdx.x;
  long long tot = (long long)Bn * N * ldk;
  if (i >= tot) return;
  int k = (int)(i % ldk); long long r = i / ldk;
  int n = (int)(r % N); int b = (int)(r / N);
  float f = (k < K) ? src[b * sstride + (long long)n * K + k] : 0.0f;
  dst[i] = f2bf(f);
}

// OIHW conv weights -> n-major [Cout][ldk], k = (ky*KW+kx)*Cin+ci.
__global__ void pack_convw_k(const float* w, unsigned short* dst, int Cout,
                             int Cin, int KH, int KW, int ldk) {
  long long i = (long long)blockIdx.x * 256 + threadIdx.x;
  long long tot = (long long)Cout * ldk;
  if (i >= tot) return;
  int k = (int)(i % ldk); int co = (int)(i / ldk);
  int K = KH * KW * Cin;
  float f = 0.0f;
  if (k < K) {
    int ci = k % Cin; int r = k / Cin;
    int kx = r % KW; int ky = r / KW;
    f = w[(((long long)co * Cin + ci) * KH + ky) * KW + kx];
  }
  dst[i] = f2bf(f);
}

// IOHW deconv weights (Cin,Cout,KH,KW), flipped -> n-major [Cout][ldk].
__global__ void pack_deconvw_k(const float* w, unsigned short* dst, int Cin,
                               int Cout, int KH, int KW, int ldk) {
  long long i = (long long)blockIdx.x * 256 + threadIdx.x;
  long long tot = (long long)Cout * ldk;
  if (i >= tot) return;
  int k = (int)(i % ldk); int co = (int)(i / ldk);
  int K = KH * KW * Cin;
  float f = 0.0f;
  if (k < K) {
    int ci = k % Cin; int r = k / Cin;
    int kx = r % KW; int ky = r / KW;
    f = w[(((long long)ci * Cout + co) * KH + (KH - 1 - ky)) * KW +
          (KW - 1 - kx)];
  }
  dst[i] = f2bf(f);
}

__global__ void add_pos_k(float* t, const float* pos, long long n, int TH) {
  long long i = (long long)blockIdx.x * 256 + threadIdx.x;
  if (i >= n) return;
  long long m = i - (i / TH) * TH;
  t[i] += pos[m];
}

__global__ void copy_cols_k(const float* src, float* dst, long long M, int C,
                            int ldsrc, int ldc, int colOff) {
  long long i = (long long)blockIdx.x * 256 + threadIdx.x;
  if (i >= M * (long long)C) return;
  long long r = i / C; int c = (int)(i - r * C);
  dst[r * ldc + colOff + c] = src[r * ldsrc + c];
}

__device__ __forceinline__ float blockReduceSum(float v) {
  __shared__ float sh[8];
  __syncthreads();
  for (int o = 16; o > 0; o >>= 1) v += __shfl_down(v, o, 32);
  if ((threadIdx.x & 31) == 0) sh[threadIdx.x >> 5] = v;
  __syncthreads();
  float r = 0.0f;
  if (threadIdx.x < 8) r = sh[threadIdx.x];
  for (int o = 4; o > 0; o >>= 1) r += __shfl_down(r, o, 32);
  if (threadIdx.x == 0) sh[0] = r;
  __syncthreads();
  return sh[0];
}

__device__ __forceinline__ float blockReduceMax(float v) {
  __shared__ float sh[8];
  __syncthreads();
  for (int o = 16; o > 0; o >>= 1) {
    float t = __shfl_down(v, o, 32); v = v > t ? v : t;
  }
  if ((threadIdx.x & 31) == 0) sh[threadIdx.x >> 5] = v;
  __syncthreads();
  float r = -3.4e38f;
  if (threadIdx.x < 8) r = sh[threadIdx.x];
  for (int o = 4; o > 0; o >>= 1) {
    float t = __shfl_down(r, o, 32); r = r > t ? r : t;
  }
  if (threadIdx.x == 0) sh[0] = r;
  __syncthreads();
  return sh[0];
}

__global__ __launch_bounds__(256)
void layernorm_k(const float* x, const float* g, const float* b, float* y,
                 int D) {
  long long row = blockIdx.x;
  const float* xr = x + row * D;
  float* yr = y + row * D;
  float s = 0.0f;
  for (int i = threadIdx.x; i < D; i += 256) s += xr[i];
  s = blockReduceSum(s);
  float mu = s / (float)D;
  float v = 0.0f;
  for (int i = threadIdx.x; i < D; i += 256) {
    float d = xr[i] - mu; v += d * d;
  }
  v = blockReduceSum(v);
  float rstd = rsqrtf(v / (float)D + 1e-12f);
  for (int i = threadIdx.x; i < D; i += 256)
    yr[i] = (xr[i] - mu) * rstd * g[i] + b[i];
}

__global__ __launch_bounds__(256)
void softmax_k(float* x, int T) {
  long long row = blockIdx.x;
  float* xr = x + row * T;
  float mx = -3.4e38f;
  for (int i = threadIdx.x; i < T; i += 256) mx = fmaxf(mx, xr[i]);
  mx = blockReduceMax(mx);
  float s = 0.0f;
  for (int i = threadIdx.x; i < T; i += 256) {
    float e = __expf(xr[i] - mx); xr[i] = e; s += e;
  }
  s = blockReduceSum(s);
  float inv = 1.0f / s;
  for (int i = threadIdx.x; i < T; i += 256) xr[i] *= inv;
}

// corr: 441 shifted dot-products over C=256 channels, /C, lrelu, BN fold.
__global__ __launch_bounds__(448)
void correlate_k(const float* a, const float* bsrc, const float* cs,
                 const float* ct, float* out, int B, int H, int W, int C,
                 int ldc, int colOff) {
  __shared__ float av[256];
  long long pix = blockIdx.x;
  int x = (int)(pix % W);
  int y = (int)((pix / W) % H);
  int b = (int)(pix / ((long long)W * H));
  const float* ap = a + pix * C;
  for (int i = threadIdx.x; i < C; i += 448) av[i] = ap[i];
  __syncthreads();
  int s = threadIdx.x;
  if (s < 441) {
    int iy = y + 2 * (s / 21) - 20;
    int ix = x + 2 * (s % 21) - 20;
    float acc = 0.0f;
    if (iy >= 0 && iy < H && ix >= 0 && ix < W) {
      const float* bp = bsrc + (((long long)b * H + iy) * W + ix) * C;
#pragma unroll 8
      for (int c = 0; c < 256; c++) acc += av[c] * bp[c];
    }
    float v = acc * (1.0f / (float)C);
    v = (v >= 0.0f) ? v : 0.1f * v;
    v = v * cs[s] + ct[s];
    out[pix * ldc + colOff + s] = v;
  }
}

// ---------------------------------------------------------------------------
// Host orchestration
// ---------------------------------------------------------------------------
extern "C" void kernel_launch(void* const* d_in, const int* in_sizes, int n_in,
                              void* d_out, int out_size, void* d_ws,
                              size_t ws_size, hipStream_t stream) {
  (void)in_sizes; (void)n_in; (void)out_size; (void)ws_size;
  auto F = [&](int i) -> const float* { return (const float*)d_in[i]; };

  char* cur = (char*)d_ws;
  auto allocB = [&](size_t bytes) -> void* {
    void* r = cur; cur += (bytes + 255) & ~(size_t)255; return r;
  };
  auto af = [&](long long n) -> float* { return (float*)allocB((size_t)n * 4); };
  auto ah = [&](long long n) -> unsigned short* {
    return (unsigned short*)allocB((size_t)n * 2);
  };
  auto cdiv = [](long long a, long long b) { return (a + b - 1) / b; };
  auto elem = [&](long long n) { return dim3((unsigned)((n + 255) / 256)); };
  auto pad32 = [](int k) { return (k + 31) & ~31; };

  const int B = 10;

  // ---- buffers (NHWC fp32 activations) ----
  float* x1n  = af((long long)B * 320 * 448 * 3);
  float* x2n  = af((long long)B * 320 * 448 * 3);
  float* c1   = af((long long)B * 160 * 224 * 64);
  float* c2a  = af((long long)B * 80 * 112 * 128);
  float* c2b  = af((long long)B * 80 * 112 * 128);
  float* c3a  = af((long long)B * 40 * 56 * 256);
  float* c3b  = af((long long)B * 40 * 56 * 256);
  float* catrc = af((long long)B * 40 * 56 * 473);
  float* oc3  = af((long long)B * 40 * 56 * 256);
  float* t1   = af((long long)B * 560 * 512);
  float* t2   = af((long long)B * 140 * 512);
  float* t3   = af((long long)B * 35 * 1024);
  float* bufA = af((long long)B * 560 * 512);
  float* bufQ = af((long long)B * 560 * 512);
  float* bufK = af((long long)B * 560 * 512);
  float* bufV = af((long long)B * 560 * 512);
  float* attb = af((long long)B * 560 * 560);
  float* bufH = af((long long)B * 560 * 2048);
  unsigned short* ktb = ah((long long)B * 560 * 1024);   // [B][T][hid]
  unsigned short* vb  = ah((long long)B * 1024 * 576);   // [B][hid][pad32(T)]
  unsigned short* wpack = ah((long long)16384 * 512);
  float* flow6 = af((long long)B * 5 * 7 * 2);
  float* flow5 = af((long long)B * 10 * 14 * 2);
  float* flow4 = af((long long)B * 20 * 28 * 2);
  float* flow3 = af((long long)B * 40 * 56 * 2);
  float* flow2 = af((long long)B * 80 * 112 * 2);
  float* cat5 = af((long long)B * 10 * 14 * 514);
  float* cat4 = af((long long)B * 20 * 28 * 770);
  float* cat3 = af((long long)B * 40 * 56 * 386);
  float* cat2 = af((long long)B * 80 * 112 * 194);

  auto gemm = [&](GemmP p, int batches) {
    dim3 g((unsigned)cdiv(p.N, 64), (unsigned)cdiv(p.M, 128),
           (unsigned)batches);
    wmma_gemm_k<<<g, dim3(256), 0, stream>>>(p);
  };

  auto bnfold = [&](int gi, int C) -> float* {  // gi -> g,b,m,v consecutive
    float* s = af(2LL * C);
    bnfold_k<<<elem(C), 256, 0, stream>>>(F(gi), F(gi + 1), F(gi + 2),
                                          F(gi + 3), s, s + C, C);
    return s;
  };

  auto conv = [&](const float* in, int Cin, int H, int W, const float* w,
                  int Cout, int KH, int st, int pd, float* out, int ldc,
                  int colOff, const float* s1, const float* sh1,
                  const float* s2, const float* sh2, const float* bias,
                  int act) {
    int Ho = (H + 2 * pd - KH) / st + 1;
    int Wo = (W + 2 * pd - KH) / st + 1;
    int ldk = pad32(KH * KH * Cin);
    pack_convw_k<<<elem((long long)Cout * ldk), 256, 0, stream>>>(
        w, wpack, Cout, Cin, KH, KH, ldk);
    GemmP p{}; p.alpha = 1.0f;
    p.A = in; p.Bm = wpack; p.C = out;
    p.M = B * Ho * Wo; p.N = Cout; p.K = KH * KH * Cin;
    p.ldb = ldk; p.ldc = ldc; p.colOff = colOff;
    p.scale1 = s1; p.shift1 = sh1; p.scale2 = s2; p.shift2 = sh2;
    p.bias = bias; p.act = act;
    p.amode = 1; p.Cin = Cin; p.H = H; p.W = W; p.KH = KH; p.KW = KH;
    p.stride = st; p.pad = pd; p.Ho = Ho; p.Wo = Wo;
    gemm(p, 1);
  };

  auto deconv = [&](const float* in, int Cin, int H, int W, const float* w,
                    int Cout, int Ho, int Wo, float* out, int ldc, int colOff,
                    int act) {
    int ldk = pad32(16 * Cin);
    pack_deconvw_k<<<elem((long long)Cout * ldk), 256, 0, stream>>>(
        w, wpack, Cin, Cout, 4, 4, ldk);
    GemmP p{}; p.alpha = 1.0f;
    p.A = in; p.Bm = wpack; p.C = out;
    p.M = B * Ho * Wo; p.N = Cout; p.K = 16 * Cin;
    p.ldb = ldk; p.ldc = ldc; p.colOff = colOff; p.act = act;
    p.amode = 2; p.Cin = Cin; p.H = H; p.W = W; p.KH = 4; p.KW = 4;
    p.stride = 1; p.pad = 2; p.Ho = Ho; p.Wo = Wo;
    gemm(p, 1);
  };

  auto linear = [&](const float* A, int M, int K, const float* w, int N,
                    const float* bias, int act, const float* resid, float* C) {
    int ldk = pad32(K);
    pack_w_tr_k<<<elem((long long)N * ldk), 256, 0, stream>>>(
        w, wpack, K, N, ldk, 0, 1);
    GemmP p{}; p.alpha = 1.0f;
    p.A = A; p.Bm = wpack; p.C = C;
    p.M = M; p.N = N; p.K = K; p.lda = K; p.ldb = ldk; p.ldc = N;
    p.bias = bias; p.act = act; p.resid = resid; p.amode = 0;
    gemm(p, 1);
  };

  // ---- input NCHW -> two NHWC images ----
  nchw_to_nhwc_k<<<elem((long long)B * 320 * 448 * 3), 256, 0, stream>>>(
      F(0), x1n, B, 6, 320, 448, 0, 3);
  nchw_to_nhwc_k<<<elem((long long)B * 320 * 448 * 3), 256, 0, stream>>>(
      F(0), x2n, B, 6, 320, 448, 3, 3);

  // ---- shared encoder (conv_block = implicit-GEMM + BN fold + lrelu) ----
  float* sb1 = bnfold(2, 64);
  float* sb2 = bnfold(7, 128);
  float* sb3 = bnfold(12, 256);
  conv(x1n, 3, 320, 448, F(1), 64, 7, 2, 3, c1, 64, 0, sb1, sb1 + 64,
       nullptr, nullptr, nullptr, 1);
  conv(c1, 64, 160, 224, F(6), 128, 5, 2, 2, c2a, 128, 0, sb2, sb2 + 128,
       nullptr, nullptr, nullptr, 1);
  conv(c2a, 128, 80, 112, F(11), 256, 5, 2, 2, c3a, 256, 0, sb3, sb3 + 256,
       nullptr, nullptr, nullptr, 1);
  conv(x2n, 3, 320, 448, F(1), 64, 7, 2, 3, c1, 64, 0, sb1, sb1 + 64,
       nullptr, nullptr, nullptr, 1);
  conv(c1, 64, 160, 224, F(6), 128, 5, 2, 2, c2b, 128, 0, sb2, sb2 + 128,
       nullptr, nullptr, nullptr, 1);
  conv(c2b, 128, 80, 112, F(11), 256, 5, 2, 2, c3b, 256, 0, sb3, sb3 + 256,
       nullptr, nullptr, nullptr, 1);

  // ---- redir (conv BN + lrelu + second BN) into concat cols [0,32) ----
  float* sr1 = bnfold(17, 32);
  float* sr2 = bnfold(21, 32);
  conv(c3a, 256, 40, 56, F(16), 32, 1, 1, 0, catrc, 473, 0, sr1, sr1 + 32,
       sr2, sr2 + 32, nullptr, 1);

  // ---- correlation (441 shifts) into concat cols [32,473) ----
  float* scorr = bnfold(25, 441);
  correlate_k<<<dim3((unsigned)(B * 40 * 56)), 448, 0, stream>>>(
      c3a, c3b, scorr, scorr + 441, catrc, B, 40, 56, 256, 473, 32);

  // ---- conv3_1 ----
  float* s31 = bnfold(30, 256);
  conv(catrc, 473, 40, 56, F(29), 256, 3, 1, 1, oc3, 256, 0, s31, s31 + 256,
       nullptr, nullptr, nullptr, 1);

  // ---- ViT stages ----
  auto vit_stage = [&](int sb, const float* vin, int Cin, int H, int W,
                       int hid, int inter, float* tout) {
    int Ho = H / 2, Wo = W / 2, T = Ho * Wo, M = B * T;
    conv(vin, Cin, H, W, F(sb + 0), hid, 2, 2, 0, tout, hid, 0, nullptr,
         nullptr, nullptr, nullptr, F(sb + 1), 0);
    add_pos_k<<<elem((long long)M * hid), 256, 0, stream>>>(
        tout, F(sb + 2), (long long)M * hid, T * hid);
    float scl = 1.0f / sqrtf((float)hid);
    int ldkH = pad32(hid);   // K-pitch for QK^T (K = hid)
    int ldkT = pad32(T);     // K-pitch for att@V (K = T)
    for (int l = 0; l < 4; l++) {
      int lb = sb + 3 + 16 * l;
      layernorm_k<<<dim3((unsigned)M), 256, 0, stream>>>(tout, F(lb + 0),
                                                         F(lb + 1), bufA, hid);
      linear(bufA, M, hid, F(lb + 2), hid, F(lb + 3), 0, nullptr, bufQ);
      linear(bufA, M, hid, F(lb + 4), hid, F(lb + 5), 0, nullptr, bufK);
      linear(bufA, M, hid, F(lb + 6), hid, F(lb + 7), 0, nullptr, bufV);
      // B(QK^T) = K^T  -> n-major [T][hid]: direct pack of K rows
      pack_w_direct_k<<<elem((long long)B * T * ldkH), 256, 0, stream>>>(
          bufK, ktb, hid, T, ldkH, (long long)T * hid, B);
      // B(att@V) = V   -> n-major [hid][T]: transpose pack of V
      pack_w_tr_k<<<elem((long long)B * hid * ldkT), 256, 0, stream>>>(
          bufV, vb, T, hid, ldkT, (long long)T * hid, B);
      {  // att = softmax(Q K^T / sqrt(hid))  (batched)
        GemmP p{}; p.alpha = scl;
        p.A = bufQ; p.Bm = ktb; p.C = attb;
        p.M = T; p.N = T; p.K = hid; p.lda = hid; p.ldb = ldkH; p.ldc = T;
        p.bAstride = (long long)T * hid;
        p.bBstride = (long long)T * ldkH;
        p.bCstride = (long long)T * T;
        gemm(p, B);
      }
      softmax_k<<<dim3((unsigned)(B * T)), 256, 0, stream>>>(attb, T);
      {  // ctx = att @ V  (batched, reuse bufQ)
        GemmP p{}; p.alpha = 1.0f;
        p.A = attb; p.Bm = vb; p.C = bufQ;
        p.M = T; p.N = hid; p.K = T; p.lda = T; p.ldb = ldkT; p.ldc = hid;
        p.bAstride = (long long)T * T;
        p.bBstride = (long long)hid * ldkT;
        p.bCstride = (long long)T * hid;
        gemm(p, B);
      }
      linear(bufQ, M, hid, F(lb + 8), hid, F(lb + 9), 0, tout, tout);
      layernorm_k<<<dim3((unsigned)M), 256, 0, stream>>>(tout, F(lb + 10),
                                                         F(lb + 11), bufA, hid);
      linear(bufA, M, hid, F(lb + 12), inter, F(lb + 13), 2, nullptr, bufH);
      linear(bufH, M, inter, F(lb + 14), hid, F(lb + 15), 0, tout, tout);
    }
  };
  vit_stage(34, oc3, 256, 40, 56, 512, 2048, t1);
  vit_stage(101, t1, 512, 20, 28, 512, 2048, t2);
  vit_stage(168, t2, 512, 10, 14, 1024, 4096, t3);

  // ---- decoder ----
  conv(t3, 1024, 5, 7, F(239), 2, 3, 1, 1, flow6, 2, 0, nullptr, nullptr,
       nullptr, nullptr, nullptr, 0);
  deconv(t3, 1024, 5, 7, F(235), 512, 10, 14, cat5, 514, 0, 1);      // d5
  deconv(flow6, 2, 5, 7, F(244), 2, 10, 14, cat5, 514, 512, 0);      // flow6_up
  conv(cat5, 514, 10, 14, F(240), 2, 3, 1, 1, flow5, 2, 0, nullptr, nullptr,
       nullptr, nullptr, nullptr, 0);
  copy_cols_k<<<elem((long long)B * 560 * 512), 256, 0, stream>>>(
      t1, cat4, (long long)B * 560, 512, 512, 770, 0);               // oc4
  deconv(cat5, 514, 10, 14, F(236), 256, 20, 28, cat4, 770, 512, 1); // d4
  deconv(flow5, 2, 10, 14, F(245), 2, 20, 28, cat4, 770, 768, 0);    // flow5_up
  conv(cat4, 770, 20, 28, F(241), 2, 3, 1, 1, flow4, 2, 0, nullptr, nullptr,
       nullptr, nullptr, nullptr, 0);
  copy_cols_k<<<elem((long long)B * 2240 * 256), 256, 0, stream>>>(
      oc3, cat3, (long long)B * 2240, 256, 256, 386, 0);             // oc3
  deconv(cat4, 770, 20, 28, F(237), 128, 40, 56, cat3, 386, 256, 1); // d3
  deconv(flow4, 2, 20, 28, F(246), 2, 40, 56, cat3, 386, 384, 0);    // flow4_up
  conv(cat3, 386, 40, 56, F(242), 2, 3, 1, 1, flow3, 2, 0, nullptr, nullptr,
       nullptr, nullptr, nullptr, 0);
  copy_cols_k<<<elem((long long)B * 8960 * 128), 256, 0, stream>>>(
      c2a, cat2, (long long)B * 8960, 128, 128, 194, 0);             // c2a
  deconv(cat3, 386, 40, 56, F(238), 64, 80, 112, cat2, 194, 128, 1); // d2
  deconv(flow3, 2, 40, 56, F(247), 2, 80, 112, cat2, 194, 192, 0);   // flow3_up
  conv(cat2, 194, 80, 112, F(243), 2, 3, 1, 1, flow2, 2, 0, nullptr, nullptr,
       nullptr, nullptr, nullptr, 0);

  nhwc_to_nchw_k<<<elem((long long)B * 2 * 80 * 112), 256, 0, stream>>>(
      flow2, (float*)d_out, B, 2, 80, 112);
}
